// SlicedMultiheadAttention_24867860644057
// MI455X (gfx1250) — matrix operand order repaired and checked by reading, not compiled, not run on hardware
//
#include <hip/hip_runtime.h>
#include <hip/hip_bf16.h>

typedef __attribute__((ext_vector_type(16))) _Float16 v16h;
typedef __attribute__((ext_vector_type(8)))  _Float16 v8h;
typedef __attribute__((ext_vector_type(8)))  float    v8f;

#define WMMA_F16(a, b, c) \
    __builtin_amdgcn_wmma_f32_16x16x32_f16(false, (a), false, (b), (short)0, (c), false, false)

// ---------------------------------------------------------------------------
// Fragment loaders (contiguous k = 16h + j order, wave32)
// A-matrix 16x32 f16: lane l=row M, half h: v16h elems j = A[M][k0+h*16+j]
// ---------------------------------------------------------------------------
__device__ __forceinline__ v16h load_a16(const _Float16* A, int lda, int row0, int k0) {
    const int lane = threadIdx.x & 31;
    const int l = lane & 15, h = lane >> 4;
    const _Float16* p = A + (size_t)(row0 + l) * lda + k0 + h * 16;
    v8h lo = *(const v8h*)p;
    v8h hi = *(const v8h*)(p + 8);
    return __builtin_shufflevector(lo, hi, 0, 1, 2, 3, 4, 5, 6, 7,
                                           8, 9, 10, 11, 12, 13, 14, 15);
}

// B-matrix 32x16 (KxN) f16: lane l = column N, half h: elems j = B[k0+h*16+j][N].
// With W stored row-major [N][K] (torch Linear weight), B[k][n] = W[n][k], so the
// 16 K-values for one lane are 16 contiguous f16 -> single 32B read.
__device__ __forceinline__ v16h load_b16(const _Float16* W, int ldw, int col0, int k0) {
    const int lane = threadIdx.x & 31;
    const int l = lane & 15, h = lane >> 4;
    return *(const v16h*)(W + (size_t)(col0 + l) * ldw + k0 + h * 16);
}

// ---------------------------------------------------------------------------
// f32 -> f16 conversion
// ---------------------------------------------------------------------------
__global__ void cvt_f32_f16(const float* __restrict__ in, _Float16* __restrict__ out, int n) {
    int i = blockIdx.x * blockDim.x + threadIdx.x;
    if (i < n) out[i] = (_Float16)in[i];
}

// ---------------------------------------------------------------------------
// GEMM: C[M,N] = (A[M,K] @ W[N,K]^T + bias[N]) * scale
// block = 256 threads (8 waves), 128x128 tile, each wave 2x4 subtiles of 16x16
// ---------------------------------------------------------------------------
__global__ __launch_bounds__(256)
void gemm_wmma_f16(const _Float16* __restrict__ A, const _Float16* __restrict__ W,
                   const float* __restrict__ bias, _Float16* __restrict__ outH,
                   float* __restrict__ outF, int M, int N, int Kd, float scale) {
    const int wave = threadIdx.x >> 5;
    const int lane = threadIdx.x & 31;
    const int l = lane & 15, h = lane >> 4;
    const int rowBase = blockIdx.y * 128 + (wave >> 1) * 32;
    const int colBase = blockIdx.x * 128 + (wave & 1) * 64;

    v8f acc[2][4];
    for (int i = 0; i < 2; ++i)
        for (int j = 0; j < 4; ++j)
            acc[i][j] = v8f{};

    for (int k0 = 0; k0 < Kd; k0 += 32) {
        v16h a0 = load_a16(A, Kd, rowBase,      k0);
        v16h a1 = load_a16(A, Kd, rowBase + 16, k0);
        v16h b0 = load_b16(W, Kd, colBase,      k0);
        v16h b1 = load_b16(W, Kd, colBase + 16, k0);
        v16h b2 = load_b16(W, Kd, colBase + 32, k0);
        v16h b3 = load_b16(W, Kd, colBase + 48, k0);
        acc[0][0] = WMMA_F16(a0, b0, acc[0][0]);
        acc[0][1] = WMMA_F16(a0, b1, acc[0][1]);
        acc[0][2] = WMMA_F16(a0, b2, acc[0][2]);
        acc[0][3] = WMMA_F16(a0, b3, acc[0][3]);
        acc[1][0] = WMMA_F16(a1, b0, acc[1][0]);
        acc[1][1] = WMMA_F16(a1, b1, acc[1][1]);
        acc[1][2] = WMMA_F16(a1, b2, acc[1][2]);
        acc[1][3] = WMMA_F16(a1, b3, acc[1][3]);
    }

    // epilogue: C layout is M = v + 8*h, N = l within each 16x16 subtile
    for (int i = 0; i < 2; ++i) {
        for (int j = 0; j < 4; ++j) {
            const int r0 = rowBase + i * 16;
            const int c0 = colBase + j * 16;
            const float bv = bias[c0 + l];
            for (int v = 0; v < 8; ++v) {
                const float val = (acc[i][j][v] + bv) * scale;
                const size_t off = (size_t)(r0 + v + 8 * h) * N + c0 + l;
                if (outH) outH[off] = (_Float16)val;
                else      outF[off] = val;
            }
        }
    }
}

// ---------------------------------------------------------------------------
// Flash attention: grid (S/64, H), block 128 threads (4 waves).
// Each wave owns 16 query rows; KV processed in steps of 32.
// Q already carries the 1/sqrt(Hd) scale (folded into the Q projection).
// ---------------------------------------------------------------------------
#define ATT_S  4096
#define ATT_D  1024
#define ATT_HD 64

__global__ __launch_bounds__(128)
void flash_attn_f16(const _Float16* __restrict__ Q, const _Float16* __restrict__ K,
                    const _Float16* __restrict__ V, _Float16* __restrict__ O) {
    const int head  = blockIdx.y;
    const int dbase = head * ATT_HD;
    const int wave  = threadIdx.x >> 5;
    const int lane  = threadIdx.x & 31;
    const int l  = lane & 15;
    const int hh = lane >> 4;
    const int qrow = blockIdx.x * 64 + wave * 16;

    __shared__ __align__(16) _Float16 Vt[ATT_HD][40];   // V^T block: [d][kv], padded stride
    __shared__ __align__(16) _Float16 Pb[4][16 * 40];   // per-wave P re-layout buffer

    // Q fragments for this wave's 16 rows (K-dim = head dim = 64 -> 2 frags)
    const v16h aq0 = load_a16(Q, ATT_D, qrow, dbase);
    const v16h aq1 = load_a16(Q, ATT_D, qrow, dbase + 32);

    v8f acc[4];
    for (int t = 0; t < 4; ++t) acc[t] = v8f{};
    float mrow[8], lrow[8];
    for (int v = 0; v < 8; ++v) { mrow[v] = -1e30f; lrow[v] = 0.0f; }

    _Float16* pb = &Pb[wave][0];

    for (int kv0 = 0; kv0 < ATT_S; kv0 += 32) {
        __syncthreads();  // protect previous iteration's Vt reads
        {   // cooperatively transpose V block (32 kv x 64 d) into Vt[d][kv]
            const int r  = threadIdx.x >> 2;        // 0..31 kv row
            const int c0 = (threadIdx.x & 3) * 16;  // d chunk
            const _Float16* vp = V + (size_t)(kv0 + r) * ATT_D + dbase + c0;
            #pragma unroll
            for (int i = 0; i < 16; ++i) Vt[c0 + i][r] = vp[i];
        }
        __syncthreads();

        // S = Q @ K^T  (two 16x16 tiles covering 32 kv columns)
        v8f s0 = v8f{}, s1 = v8f{};
        s0 = WMMA_F16(aq0, load_b16(K, ATT_D, kv0,      dbase),      s0);
        s0 = WMMA_F16(aq1, load_b16(K, ATT_D, kv0,      dbase + 32), s0);
        s1 = WMMA_F16(aq0, load_b16(K, ATT_D, kv0 + 16, dbase),      s1);
        s1 = WMMA_F16(aq1, load_b16(K, ATT_D, kv0 + 16, dbase + 32), s1);

        // online softmax: row m = v + 8*hh lives in one 16-lane half
        #pragma unroll
        for (int v = 0; v < 8; ++v) {
            float m = fmaxf(s0[v], s1[v]);
            m = fmaxf(m, __shfl_xor(m, 1, 32));
            m = fmaxf(m, __shfl_xor(m, 2, 32));
            m = fmaxf(m, __shfl_xor(m, 4, 32));
            m = fmaxf(m, __shfl_xor(m, 8, 32));
            const float mnew  = fmaxf(mrow[v], m);
            const float alpha = __expf(mrow[v] - mnew);
            mrow[v] = mnew;
            const float p0 = __expf(s0[v] - mnew);
            const float p1 = __expf(s1[v] - mnew);
            s0[v] = p0; s1[v] = p1;
            float rs = p0 + p1;
            rs += __shfl_xor(rs, 1, 32);
            rs += __shfl_xor(rs, 2, 32);
            rs += __shfl_xor(rs, 4, 32);
            rs += __shfl_xor(rs, 8, 32);
            lrow[v] = lrow[v] * alpha + rs;
            acc[0][v] *= alpha; acc[1][v] *= alpha;
            acc[2][v] *= alpha; acc[3][v] *= alpha;
        }

        // P: C-layout -> LDS -> A-layout (wave-private buffer, stride 40)
        #pragma unroll
        for (int v = 0; v < 8; ++v) {
            pb[(v + 8 * hh) * 40 + l]      = (_Float16)s0[v];
            pb[(v + 8 * hh) * 40 + 16 + l] = (_Float16)s1[v];
        }
        asm volatile("s_wait_dscnt 0" ::: "memory");  // wave-local LDS RAW
        const v16h ap = load_a16(pb, 40, 0, 0);

        // O += P(16x32) @ V(32x64): B-frag elems j = V[kv0+hh*16+j][d0+l] = Vt[d0+l][hh*16+j]
        #pragma unroll
        for (int t = 0; t < 4; ++t) {
            const v16h bv = *(const v16h*)(&Vt[t * 16 + l][hh * 16]);
            acc[t] = WMMA_F16(ap, bv, acc[t]);
        }
    }

    // normalize and store context (f16), layout [s][head*64 + d]
    #pragma unroll
    for (int t = 0; t < 4; ++t) {
        #pragma unroll
        for (int v = 0; v < 8; ++v) {
            const float val = acc[t][v] / lrow[v];
            O[(size_t)(qrow + v + 8 * hh) * ATT_D + dbase + t * 16 + l] = (_Float16)val;
        }
    }
}

// ---------------------------------------------------------------------------
// host launcher
// ---------------------------------------------------------------------------
extern "C" void kernel_launch(void* const* d_in, const int* in_sizes, int n_in,
                              void* d_out, int out_size, void* d_ws, size_t ws_size,
                              hipStream_t stream) {
    (void)in_sizes; (void)n_in; (void)out_size; (void)ws_size;
    const int S = 4096, D = 1024;
    const float SCALE = 0.125f;  // (D/H)^-0.5 = 64^-0.5

    const float* x_q = (const float*)d_in[0];
    const float* x_k = (const float*)d_in[1];
    const float* x_v = (const float*)d_in[2];
    const float* Wq  = (const float*)d_in[3];
    const float* bq  = (const float*)d_in[4];
    const float* Wk  = (const float*)d_in[5];
    const float* bk  = (const float*)d_in[6];
    const float* Wv  = (const float*)d_in[7];
    const float* bv  = (const float*)d_in[8];
    const float* Wo  = (const float*)d_in[9];
    const float* bo  = (const float*)d_in[10];

    const size_t SD = (size_t)S * D;   // 4M elems
    const size_t DD = (size_t)D * D;   // 1M elems
    _Float16* xqh = (_Float16*)d_ws;
    _Float16* xkh = xqh + SD;
    _Float16* xvh = xkh + SD;
    _Float16* wqh = xvh + SD;
    _Float16* wkh = wqh + DD;
    _Float16* wvh = wkh + DD;
    _Float16* woh = wvh + DD;
    _Float16* qh  = woh + DD;
    _Float16* kh  = qh  + SD;
    _Float16* vh  = kh  + SD;
    _Float16* ctx = vh  + SD;          // total ws = (7*SD + 4*DD)*2 = 64 MB

    const int cb = 256;
    cvt_f32_f16<<<(int)((SD + cb - 1) / cb), cb, 0, stream>>>(x_q, xqh, (int)SD);
    cvt_f32_f16<<<(int)((SD + cb - 1) / cb), cb, 0, stream>>>(x_k, xkh, (int)SD);
    cvt_f32_f16<<<(int)((SD + cb - 1) / cb), cb, 0, stream>>>(x_v, xvh, (int)SD);
    cvt_f32_f16<<<(int)((DD + cb - 1) / cb), cb, 0, stream>>>(Wq, wqh, (int)DD);
    cvt_f32_f16<<<(int)((DD + cb - 1) / cb), cb, 0, stream>>>(Wk, wkh, (int)DD);
    cvt_f32_f16<<<(int)((DD + cb - 1) / cb), cb, 0, stream>>>(Wv, wvh, (int)DD);
    cvt_f32_f16<<<(int)((DD + cb - 1) / cb), cb, 0, stream>>>(Wo, woh, (int)DD);

    // projections: q gets the softmax scale folded in
    dim3 gProj(D / 128, S / 128);  // (8, 32)
    gemm_wmma_f16<<<gProj, 256, 0, stream>>>(xqh, wqh, bq, qh, nullptr, S, D, D, SCALE);
    gemm_wmma_f16<<<gProj, 256, 0, stream>>>(xkh, wkh, bk, kh, nullptr, S, D, D, 1.0f);
    gemm_wmma_f16<<<gProj, 256, 0, stream>>>(xvh, wvh, bv, vh, nullptr, S, D, D, 1.0f);

    // attention
    dim3 gAttn(S / 64, 16);
    flash_attn_f16<<<gAttn, 128, 0, stream>>>(qh, kh, vh, ctx);

    // output projection -> f32 result
    gemm_wmma_f16<<<gProj, 256, 0, stream>>>(ctx, woh, bo, nullptr, (float*)d_out, S, D, D, 1.0f);
}
